// QWenAttention_26568667693283
// MI455X (gfx1250) — matrix operand order repaired and checked by reading, not compiled, not run on hardware
//
#include <hip/hip_runtime.h>
#include <hip/hip_bf16.h>

// ---------------------------------------------------------------------------
// QWen-style attention layer for MI455X (gfx1250), wave32 + WMMA bf16.
//   B=1, S=2048, H=2048, NH=16, HD=128, rope base 10000
// Pipeline: convert -> QKV GEMM(+bias+RoPE) -> flash attention -> proj GEMM
// Global->LDS staging uses CDNA5 async-to-LDS DMA (ASYNCcnt) where the copy
// is transpose-free; fragments stream from LDS via ds_load_b128.
// ---------------------------------------------------------------------------

typedef __attribute__((ext_vector_type(16))) __bf16 v16bf;
typedef __attribute__((ext_vector_type(8)))  float  v8f;

#define S_LEN 2048
#define HID   2048
#define NHEAD 16
#define HDIM  128

// ----- CDNA5 async global->LDS copy (16B per lane), tracked by ASYNCcnt -----
__device__ __forceinline__ void async_copy_b128(void* lds, const void* g)
{
    asm volatile("global_load_async_to_lds_b128 %0, %1, off"
                 :: "v"((unsigned int)(unsigned long long)lds), "v"(g)
                 : "memory");
}
__device__ __forceinline__ void wait_async()
{
    asm volatile("s_wait_asynccnt 0x0" ::: "memory");
}

// ---------------------------------------------------------------------------
// Kernel 0: f32 -> bf16 convert (weights become L2-resident bf16)
// ---------------------------------------------------------------------------
__global__ __launch_bounds__(256) void f32_to_bf16_kernel(
    const float* __restrict__ in, __bf16* __restrict__ out, int n)
{
    int i = blockIdx.x * blockDim.x + threadIdx.x;
    int stride = gridDim.x * blockDim.x;
    for (; i < n; i += stride) out[i] = (__bf16)in[i];
}

// ---------------------------------------------------------------------------
// GEMM: C[M,N] = A[M,K](bf16) * B[K,N](bf16)  via v_wmma_f32_16x16x32_bf16
// workgroup tile 128x128, 8 waves as 2(M)x4(N), wave tile 64x32 (4x2 accums)
// MODE 0: store f32 to outF
// MODE 1: +bias, RoPE on Q/K, scatter to q[H][S][D], k[H][S][D], v_t[H][D][S]
// ---------------------------------------------------------------------------
template<int MODE>
__global__ __launch_bounds__(256) void gemm128_wmma(
    const __bf16* __restrict__ A, const __bf16* __restrict__ Bw,
    const float* __restrict__ bias, float* __restrict__ outF,
    __bf16* __restrict__ outQ, __bf16* __restrict__ outK, __bf16* __restrict__ outV,
    int M, int N, int K)
{
    __shared__ __bf16 As[128][40];   // [m][k], +8 pad, rows 80B (16B-aligned chunks)
    __shared__ __bf16 Bs[128][40];   // [n][k] (transposed), B-frag = contiguous k

    const int t     = threadIdx.x;
    const int wave  = t >> 5;
    const int lane  = t & 31;
    const int waveM = wave & 1;       // 0..1 -> 64 rows each
    const int waveN = wave >> 1;      // 0..3 -> 32 cols each
    const int g     = lane >> 4;      // half-wave select
    const int ln    = lane & 15;
    const int m0    = blockIdx.y * 128;
    const int n0    = blockIdx.x * 128;

    const v8f vz = {0.f,0.f,0.f,0.f,0.f,0.f,0.f,0.f};
    v8f acc[4][2];
#pragma unroll
    for (int tm = 0; tm < 4; ++tm)
#pragma unroll
        for (int tn = 0; tn < 2; ++tn) acc[tm][tn] = vz;

    const int arow  = t >> 2;         // 0..63 (+64*p), A staging row
    const int acol8 = (t & 3) * 8;    // 0,8,16,24  (16B-aligned chunk)
    const int brow  = t >> 5;         // 0..7  (+8*p), B staging row
    const int bcol  = (t & 31) * 4;   // 0..124

    for (int k0 = 0; k0 < K; k0 += 32) {
        __syncthreads();
        // stage A tile 128x32 via async DMA (straight copy, no transpose)
#pragma unroll
        for (int p = 0; p < 2; ++p) {
            int r = arow + 64 * p;
            async_copy_b128(&As[r][acol8],
                            A + (size_t)(m0 + r) * K + k0 + acol8);
        }
        // stage B tile 32x128 transposed -> Bs[n][k] (manual, needs transpose)
#pragma unroll
        for (int p = 0; p < 4; ++p) {
            int r = brow + 8 * p;
            ushort4 v = *(const ushort4*)(Bw + (size_t)(k0 + r) * N + n0 + bcol);
            union { ushort4 u; __bf16 h[4]; } cv; cv.u = v;
            Bs[bcol + 0][r] = cv.h[0];
            Bs[bcol + 1][r] = cv.h[1];
            Bs[bcol + 2][r] = cv.h[2];
            Bs[bcol + 3][r] = cv.h[3];
        }
        if (k0 + 32 < K) {  // prefetch next tiles -> global_prefetch_b8
            __builtin_prefetch(A + (size_t)(m0 + arow) * K + k0 + 32 + acol8, 0, 3);
            __builtin_prefetch(Bw + (size_t)(k0 + 32 + brow) * N + n0 + bcol, 0, 3);
        }
        wait_async();
        __syncthreads();

        // A fragments: lane m = ln; halves 0-7 <- K=8g+0..7, 8-15 <- K=16+8g+0..7
        v16bf af[4];
#pragma unroll
        for (int tm = 0; tm < 4; ++tm) {
            const __bf16* pa = &As[waveM * 64 + tm * 16 + ln][8 * g];
            uint4* d = (uint4*)&af[tm];
            d[0] = ((const uint4*)pa)[0];
            d[1] = *(const uint4*)(pa + 16);
        }
        // B fragments: lane n = ln; halves 0-7 <- K=8g+0..7, 8-15 <- K=16+8g+0..7
        v16bf bf[2];
#pragma unroll
        for (int tn = 0; tn < 2; ++tn) {
            const __bf16* pb = &Bs[waveN * 32 + tn * 16 + ln][8 * g];
            uint4* d = (uint4*)&bf[tn];
            d[0] = ((const uint4*)pb)[0];
            d[1] = *(const uint4*)(pb + 16);
        }
#pragma unroll
        for (int tm = 0; tm < 4; ++tm)
#pragma unroll
            for (int tn = 0; tn < 2; ++tn)
                acc[tm][tn] = __builtin_amdgcn_wmma_f32_16x16x32_bf16(
                    false, af[tm], false, bf[tn], (short)0, acc[tm][tn], false, false);
    }

    if constexpr (MODE == 0) {
        // plain f32 store (C layout: VGPR r -> row r+8g, col = ln)
#pragma unroll
        for (int tm = 0; tm < 4; ++tm)
#pragma unroll
            for (int tn = 0; tn < 2; ++tn)
#pragma unroll
                for (int r = 0; r < 8; ++r) {
                    int row = m0 + waveM * 64 + tm * 16 + r + 8 * g;
                    int col = n0 + waveN * 32 + tn * 16 + ln;
                    outF[(size_t)row * N + col] = acc[tm][tn][r];
                }
    } else {
        // bias + RoPE epilogue through LDS (two 64-row phases, < 64KB static LDS)
        __shared__ float Cs[64][132];
        const int nb = blockIdx.x;   // 0..15 Q-head, 16..31 K-head, 32..47 V-head
#pragma unroll 1
        for (int ph = 0; ph < 2; ++ph) {
            __syncthreads();
            if (waveM == ph) {
#pragma unroll
                for (int tm = 0; tm < 4; ++tm)
#pragma unroll
                    for (int tn = 0; tn < 2; ++tn)
#pragma unroll
                        for (int r = 0; r < 8; ++r) {
                            int rl = tm * 16 + r + 8 * g;
                            int cl = waveN * 32 + tn * 16 + ln;
                            Cs[rl][cl] = acc[tm][tn][r] + bias[n0 + cl];
                        }
            }
            __syncthreads();
            const int rl = t >> 2;           // 0..63
            const int dq = (t & 3) * 32;     // 0,32,64,96
            const int sg = m0 + ph * 64 + rl;
            if (nb < 32) {
                const int head = nb & 15;
                __bf16* dst = (nb < 16) ? outQ : outK;
#pragma unroll
                for (int j = 0; j < 32; ++j) {
                    int d  = dq + j;
                    int jj = d & 63;
                    float x = Cs[rl][d];
                    float partner = (d < 64) ? -Cs[rl][d + 64] : Cs[rl][d - 64];
                    float invf = __powf(10000.0f, (float)(-2 * jj) * 0.0078125f);
                    float ang  = (float)sg * invf;
                    float c, sn;
                    __sincosf(ang, &sn, &c);
                    dst[((size_t)head * S_LEN + sg) * HDIM + d] = (__bf16)(x * c + partner * sn);
                }
            } else {
                const int head = nb - 32;    // V, store transposed [H][D][S]
#pragma unroll
                for (int j = 0; j < 32; ++j) {
                    int d = dq + j;
                    outV[((size_t)head * HDIM + d) * S_LEN + sg] = (__bf16)Cs[rl][d];
                }
            }
        }
    }
}

// ---------------------------------------------------------------------------
// Flash attention: one workgroup per (head, 128-row q tile); k-blocks of 64.
// Each wave owns 16 q rows; scores via wmma(Q,K^T), online softmax in f32,
// P -> LDS (per-wave region) -> A-fragments -> wmma(P,V).
// K/V tiles staged by async global->LDS DMA.
// ---------------------------------------------------------------------------
__global__ __launch_bounds__(256) void flash_attn_wmma(
    const __bf16* __restrict__ q, const __bf16* __restrict__ k,
    const __bf16* __restrict__ vt, __bf16* __restrict__ ctx)
{
    const int qb   = blockIdx.x;   // 0..15
    const int head = blockIdx.y;   // 0..15
    const int t    = threadIdx.x;
    const int wave = t >> 5;
    const int lane = t & 31;
    const int g    = lane >> 4;
    const int ln   = lane & 15;

    __shared__ __bf16 Ks[64][72];    // [k_seq][d]
    __shared__ __bf16 Vs[128][72];   // [d][k_seq]
    __shared__ __bf16 Ps[128][72];   // per-wave 16-row regions

    // preload Q A-fragments directly from global (read once, interleaved layout)
    const int qrow = qb * 128 + wave * 16 + ln;
    const __bf16* qp = q + ((size_t)head * S_LEN + qrow) * HDIM;
    v16bf qf[4];
#pragma unroll
    for (int ks = 0; ks < 4; ++ks) {
        uint4* d = (uint4*)&qf[ks];
        d[0] = *(const uint4*)(qp + 32 * ks + 8 * g);
        d[1] = *(const uint4*)(qp + 32 * ks + 16 + 8 * g);
    }

    const v8f vz = {0.f,0.f,0.f,0.f,0.f,0.f,0.f,0.f};
    v8f o[8];
#pragma unroll
    for (int i = 0; i < 8; ++i) o[i] = vz;
    float mi[8], li[8];
#pragma unroll
    for (int r = 0; r < 8; ++r) { mi[r] = -3.0e38f; li[r] = 0.f; }

    const float scale = 0.08838834764831845f;  // 1/sqrt(128)
    const int kb_max = 2 * qb + 1;

    for (int kb = 0; kb <= kb_max; ++kb) {
        __syncthreads();
        {   // stage K tile 64x128 via async DMA (4 x 16B per thread)
            int krow = t >> 2, qtr = t & 3;
            const __bf16* src = k + ((size_t)head * S_LEN + kb * 64 + krow) * HDIM + qtr * 32;
            __bf16* dst = &Ks[krow][qtr * 32];
            async_copy_b128(dst,      src);
            async_copy_b128(dst + 8,  src + 8);
            async_copy_b128(dst + 16, src + 16);
            async_copy_b128(dst + 24, src + 24);
        }
        {   // stage V^T tile 128(d) x 64(k) via async DMA
            int d = t >> 1, half = t & 1;
            const __bf16* src = vt + ((size_t)head * HDIM + d) * S_LEN + kb * 64 + half * 32;
            __bf16* dst = &Vs[d][half * 32];
            async_copy_b128(dst,      src);
            async_copy_b128(dst + 8,  src + 8);
            async_copy_b128(dst + 16, src + 16);
            async_copy_b128(dst + 24, src + 24);
        }
        wait_async();
        __syncthreads();

        // scores: wave computes 16(q) x 64(k), K-dim = d = 128 in 4 steps;
        // batch the 4 K-fragments per step so DS loads overlap WMMA issue
        v8f s[4];
#pragma unroll
        for (int tn = 0; tn < 4; ++tn) s[tn] = vz;
#pragma unroll
        for (int ks = 0; ks < 4; ++ks) {
            v16bf bk[4];
#pragma unroll
            for (int tn = 0; tn < 4; ++tn) {
                const __bf16* kp = &Ks[tn * 16 + ln][32 * ks + 8 * g];
                uint4* bd = (uint4*)&bk[tn];
                bd[0] = ((const uint4*)kp)[0];
                bd[1] = *(const uint4*)(kp + 16);
            }
#pragma unroll
            for (int tn = 0; tn < 4; ++tn)
                s[tn] = __builtin_amdgcn_wmma_f32_16x16x32_bf16(
                    false, qf[ks], false, bk[tn], (short)0, s[tn], false, false);
        }

        // online softmax (rows live in 16-lane halves; shfl_xor width 16)
        const bool maskblk = (kb >= 2 * qb);
        const int qbase = qb * 128 + wave * 16;
#pragma unroll
        for (int r = 0; r < 8; ++r) {
            const int qg = qbase + r + 8 * g;
            float rowm = -3.0e38f;
#pragma unroll
            for (int tn = 0; tn < 4; ++tn) {
                float v = s[tn][r] * scale;
                int kg = kb * 64 + tn * 16 + ln;
                if (maskblk && kg > qg) v = -3.0e38f;
                s[tn][r] = v;
                rowm = fmaxf(rowm, v);
            }
            rowm = fmaxf(rowm, __shfl_xor(rowm, 1, 16));
            rowm = fmaxf(rowm, __shfl_xor(rowm, 2, 16));
            rowm = fmaxf(rowm, __shfl_xor(rowm, 4, 16));
            rowm = fmaxf(rowm, __shfl_xor(rowm, 8, 16));
            float mnew  = fmaxf(mi[r], rowm);
            float alpha = __expf(mi[r] - mnew);
            float rs = 0.f;
#pragma unroll
            for (int tn = 0; tn < 4; ++tn) {
                float p = __expf(s[tn][r] - mnew);
                s[tn][r] = p;
                rs += p;
            }
            rs += __shfl_xor(rs, 1, 16);
            rs += __shfl_xor(rs, 2, 16);
            rs += __shfl_xor(rs, 4, 16);
            rs += __shfl_xor(rs, 8, 16);
            li[r] = li[r] * alpha + rs;
            mi[r] = mnew;
#pragma unroll
            for (int tnd = 0; tnd < 8; ++tnd) o[tnd][r] *= alpha;
        }

        // P (C-layout) -> per-wave LDS region -> A-fragment layout
#pragma unroll
        for (int tn = 0; tn < 4; ++tn)
#pragma unroll
            for (int r = 0; r < 8; ++r)
                Ps[wave * 16 + r + 8 * g][tn * 16 + ln] = (__bf16)s[tn][r];

        v16bf pf[2];
#pragma unroll
        for (int ks2 = 0; ks2 < 2; ++ks2) {
            const __bf16* pp = &Ps[wave * 16 + ln][32 * ks2 + 8 * g];
            uint4* d = (uint4*)&pf[ks2];
            d[0] = ((const uint4*)pp)[0];
            d[1] = *(const uint4*)(pp + 16);
        }

        // O += P @ V : n = d (8 tiles), K-dim = k_seq = 64 in 2 steps;
        // batch the 8 V-fragments per step
#pragma unroll
        for (int ks2 = 0; ks2 < 2; ++ks2) {
            v16bf vf[8];
#pragma unroll
            for (int tnd = 0; tnd < 8; ++tnd) {
                const __bf16* vp = &Vs[tnd * 16 + ln][32 * ks2 + 8 * g];
                uint4* vd = (uint4*)&vf[tnd];
                vd[0] = ((const uint4*)vp)[0];
                vd[1] = *(const uint4*)(vp + 16);
            }
#pragma unroll
            for (int tnd = 0; tnd < 8; ++tnd)
                o[tnd] = __builtin_amdgcn_wmma_f32_16x16x32_bf16(
                    false, pf[ks2], false, vf[tnd], (short)0, o[tnd], false, false);
        }
    }

    // normalize and store ctx[s][H] as bf16 for the projection GEMM
#pragma unroll
    for (int tnd = 0; tnd < 8; ++tnd)
#pragma unroll
        for (int r = 0; r < 8; ++r) {
            int row = qb * 128 + wave * 16 + r + 8 * g;
            int d   = tnd * 16 + ln;
            float val = o[tnd][r] / li[r];
            ctx[(size_t)row * HID + head * HDIM + d] = (__bf16)val;
        }
}

// ---------------------------------------------------------------------------
// Launch
// ---------------------------------------------------------------------------
extern "C" void kernel_launch(void* const* d_in, const int* in_sizes, int n_in,
                              void* d_out, int out_size, void* d_ws, size_t ws_size,
                              hipStream_t stream)
{
    const float* hidden = (const float*)d_in[0];  // [1,2048,2048]
    const float* w_qkv  = (const float*)d_in[1];  // [2048,6144]
    const float* b_qkv  = (const float*)d_in[2];  // [6144]
    const float* w_proj = (const float*)d_in[3];  // [2048,2048]
    float* out = (float*)d_out;                   // [1,2048,2048]

    char* ws = (char*)d_ws;
    __bf16* hb     = (__bf16*)(ws + 0);                    //  8 MiB  hidden bf16
    __bf16* wqkvb  = (__bf16*)(ws + 8388608ull);           // 24 MiB  w_qkv bf16
    __bf16* wprojb = (__bf16*)(ws + 33554432ull);          //  8 MiB  w_proj bf16
    __bf16* qr     = (__bf16*)(ws + 41943040ull);          //  8 MiB  q rope [H][S][D]
    __bf16* kr     = (__bf16*)(ws + 50331648ull);          //  8 MiB  k rope [H][S][D]
    __bf16* vtb    = (__bf16*)(ws + 58720256ull);          //  8 MiB  v^T   [H][D][S]
    __bf16* ctxb   = (__bf16*)(ws + 67108864ull);          //  8 MiB  ctx   [S][H]

    f32_to_bf16_kernel<<<4096, 256, 0, stream>>>(hidden, hb,     HID * S_LEN);
    f32_to_bf16_kernel<<<8192, 256, 0, stream>>>(w_qkv,  wqkvb,  HID * 3 * HID);
    f32_to_bf16_kernel<<<4096, 256, 0, stream>>>(w_proj, wprojb, HID * HID);

    // QKV GEMM + bias + RoPE scatter: grid (N/128=48, M/128=16)
    dim3 g1(48, 16);
    gemm128_wmma<1><<<g1, 256, 0, stream>>>(hb, wqkvb, b_qkv, nullptr,
                                            qr, kr, vtb, S_LEN, 3 * HID, HID);

    // Flash attention: grid (S/128=16 q-tiles, 16 heads)
    dim3 g2(16, 16);
    flash_attn_wmma<<<g2, 256, 0, stream>>>(qr, kr, vtb, ctxb);

    // Output projection: grid (16, 16)
    dim3 g3(16, 16);
    gemm128_wmma<0><<<g3, 256, 0, stream>>>(ctxb, wprojb, nullptr, out,
                                            nullptr, nullptr, nullptr, S_LEN, HID, HID);
}
